// CoAttention_60206851555510
// MI455X (gfx1250) — compile-verified, matrix operand order checked by reading, not run on hardware
//
#include <hip/hip_runtime.h>
#include <math.h>

// ---------------------------------------------------------------------------
// CoAttention, mathematically collapsed (rank-1 score structure => softmax of
// a sum with a term constant along the softmax axis drops that term exactly).
// ---------------------------------------------------------------------------

#define B_DIM 32
#define S_DIM 512
#define R_DIM 196
#define D_DIM 768

typedef __attribute__((ext_vector_type(16))) __bf16 v16bf;
typedef __attribute__((ext_vector_type(8)))  __bf16 v8bf;
typedef __attribute__((ext_vector_type(8)))  float  v8f;

// dynamic LDS layout for the GEMM kernel (no static __shared__ in kernel, so
// extern-shared base == LDS offset 0, matching raw offsets in the async asm):
//   [0   ..  64)   rowsum[16] (f32)
//   [128 .. 128+48K)        B-stage buffer 0  (768 cols x 32 K bf16)
//   [128+48K .. 128+96K)    B-stage buffer 1
#define LDS_ROWSUM   0
#define LDS_BSTAGE   128
#define BSTAGE_BYTES 49152           // 768*32*2
#define GEMM_LDS_BYTES (LDS_BSTAGE + 2 * BSTAGE_BYTES)

// ---------------------------------------------------------------------------
// Kernel 1: convert + transpose weight  WT[n][k] = bf16(W[k][n]),  D x D
// ---------------------------------------------------------------------------
__global__ __launch_bounds__(256) void k_weightT(const float* __restrict__ W,
                                                 __bf16* __restrict__ WT) {
  int idx = blockIdx.x * 256 + threadIdx.x;          // idx = n*D + k
  if (idx >= D_DIM * D_DIM) return;
  int n = idx / D_DIM;
  int k = idx - n * D_DIM;
  WT[idx] = (__bf16)W[(size_t)k * D_DIM + n];
}

// ---------------------------------------------------------------------------
// Kernel 2: fused GEMM + tanh + dot-with-wvec row reduction.
//   out[m] = sum_n tanh( (A @ W)[m,n] ) * wvec[n]
// A: (M,768) f32 row-major.  WT: bf16 W^T, WT[n][k].  wvec: 768 f32.
// Block = 256 threads (8 waves), owns a 16-row M tile; wave w covers columns
// { w*16 + j*128 + (0..15) : j=0..5 } => all 768 columns per block.
// B K-chunks (48KB) are double-buffered in LDS via global_load_async_to_lds,
// overlapping DMA with the bf16 WMMA + tanh compute of the previous chunk.
// ---------------------------------------------------------------------------
__device__ __forceinline__ void issue_b_chunk(const __bf16* __restrict__ WT,
                                              int k0, unsigned ldsbase, int tid) {
#pragma unroll
  for (int i = 0; i < 12; ++i) {
    const int c    = i * 256 + tid;        // 0..3071 16-byte chunks
    const int n    = c >> 2;               // column 0..767
    const int part = c & 3;                // 16B slice within 64B row
    const unsigned lds = ldsbase + (unsigned)(n * 64 + part * 16);
    const char* g = (const char*)(WT + (size_t)n * D_DIM + k0) + part * 16;
    asm volatile("global_load_async_to_lds_b128 %0, %1, off"
                 :: "v"(lds), "v"((unsigned long long)(uintptr_t)g)
                 : "memory");
  }
}

__global__ __launch_bounds__(256)
void k_gemm_tanh_dot(const float* __restrict__ A,
                     const __bf16* __restrict__ WT,
                     const float* __restrict__ wvec,
                     float* __restrict__ out, int M) {
  extern __shared__ char smem[];
  float* rowsum = (float*)(smem + LDS_ROWSUM);

  const int tid   = threadIdx.x;
  const int wave  = tid >> 5;
  const int lane  = tid & 31;
  const int mloc  = lane & 15;           // M within tile (A fragment row)
  const int khalf = (lane >> 4) * 8;     // K sub-offset per ISA 16-bit layout
  const int mbase = blockIdx.x * 16;
  if (mbase >= M) return;

  if (tid < 16) rowsum[tid] = 0.0f;

  const float* Arow = A + (size_t)(mbase + mloc) * D_DIM + khalf;

  v8f acc[6] = {};   // six 16x16 fp32 accumulator tiles per wave

  // prologue: start DMA of chunk 0 into buffer 0
  issue_b_chunk(WT, 0, LDS_BSTAGE, tid);

  for (int i = 0; i < D_DIM / 32; ++i) {           // 24 K-chunks
    const int k0 = i * 32;
    // wait for this wave's chunk-i async copies, then block-wide visibility.
    asm volatile("s_wait_asynccnt 0x0" ::: "memory");
    __syncthreads();
    // kick off DMA for the next chunk into the other buffer (overlaps WMMA).
    if (i + 1 < D_DIM / 32)
      issue_b_chunk(WT, k0 + 32, LDS_BSTAGE + ((i + 1) & 1) * BSTAGE_BYTES, tid);

    // ---- A fragment: 16-bit A 16x32 layout, f32 -> bf16 in-register.
    const float* ap = Arow + k0;
    __builtin_prefetch(ap + 32, 0, 3);             // prefetch next A chunk
    const float4 f0 = *reinterpret_cast<const float4*>(ap);
    const float4 f1 = *reinterpret_cast<const float4*>(ap + 4);
    const float4 f2 = *reinterpret_cast<const float4*>(ap + 16);
    const float4 f3 = *reinterpret_cast<const float4*>(ap + 20);
    v16bf a;
    a[0]  = (__bf16)f0.x; a[1]  = (__bf16)f0.y; a[2]  = (__bf16)f0.z; a[3]  = (__bf16)f0.w;
    a[4]  = (__bf16)f1.x; a[5]  = (__bf16)f1.y; a[6]  = (__bf16)f1.z; a[7]  = (__bf16)f1.w;
    a[8]  = (__bf16)f2.x; a[9]  = (__bf16)f2.y; a[10] = (__bf16)f2.z; a[11] = (__bf16)f2.w;
    a[12] = (__bf16)f3.x; a[13] = (__bf16)f3.y; a[14] = (__bf16)f3.z; a[15] = (__bf16)f3.w;

    const char* bbuf = smem + LDS_BSTAGE + (i & 1) * BSTAGE_BYTES;
#pragma unroll
    for (int j = 0; j < 6; ++j) {
      const int n = wave * 16 + j * 128 + (lane & 15);
      const char* bp = bbuf + n * 64 + khalf * 2;
      const v8bf blo = *reinterpret_cast<const v8bf*>(bp);        // ds_load_b128
      const v8bf bhi = *reinterpret_cast<const v8bf*>(bp + 32);   // K+16..23
      const v16bf b = __builtin_shufflevector(blo, bhi,
          0, 1, 2, 3, 4, 5, 6, 7, 8, 9, 10, 11, 12, 13, 14, 15);
      acc[j] = __builtin_amdgcn_wmma_f32_16x16x32_bf16(
          false, a, false, b, (short)0, acc[j], false, false);
    }
  }

  // ---- epilogue: tanh, scale by wvec[n], reduce over N into 16 row sums.
#pragma unroll
  for (int j = 0; j < 6; ++j) {
    const int n = wave * 16 + j * 128 + (lane & 15);
    const float wv = wvec[n];
#pragma unroll
    for (int e = 0; e < 8; ++e) {
      float v = tanhf(acc[j][e]) * wv;
      // butterfly across the 16 lanes of this half (masks stay within half)
      v += __shfl_xor(v, 1, 32);
      v += __shfl_xor(v, 2, 32);
      v += __shfl_xor(v, 4, 32);
      v += __shfl_xor(v, 8, 32);
      if ((lane & 15) == 0) {
        const int m = e + ((lane >> 4) << 3);  // C layout: M = e + 8*(lane/16)
        atomicAdd(&rowsum[m], v);              // ds_add_f32
      }
    }
  }
  __syncthreads();
  if (tid < 16) out[mbase + tid] = rowsum[tid];
}

// ---------------------------------------------------------------------------
// Kernel 3: per-batch softmax + weighted pooling.
//   v1[b,:] = softmax(si[b,:]) @ img[b]   (R = 196)
//   v2[b,:] = softmax(kt[b,:]) @ text[b]  (S = 512)
// ---------------------------------------------------------------------------
__device__ __forceinline__ float blk_max(float v, float* red, int tid) {
#pragma unroll
  for (int m = 16; m >= 1; m >>= 1) v = fmaxf(v, __shfl_xor(v, m, 32));
  if ((tid & 31) == 0) red[tid >> 5] = v;
  __syncthreads();
  float r = fmaxf(fmaxf(fmaxf(red[0], red[1]), fmaxf(red[2], red[3])),
                  fmaxf(fmaxf(red[4], red[5]), fmaxf(red[6], red[7])));
  __syncthreads();
  return r;
}
__device__ __forceinline__ float blk_sum(float v, float* red, int tid) {
#pragma unroll
  for (int m = 16; m >= 1; m >>= 1) v += __shfl_xor(v, m, 32);
  if ((tid & 31) == 0) red[tid >> 5] = v;
  __syncthreads();
  float r = (red[0] + red[1]) + (red[2] + red[3]) +
            (red[4] + red[5]) + (red[6] + red[7]);
  __syncthreads();
  return r;
}

__global__ __launch_bounds__(256)
void k_attend_pool(const float* __restrict__ si, const float* __restrict__ kt,
                   const float* __restrict__ img, const float* __restrict__ text,
                   float* __restrict__ v1, float* __restrict__ v2) {
  __shared__ float p[S_DIM];
  __shared__ float red[8];
  const int b = blockIdx.x, tid = threadIdx.x;

  // ---- visual attention over R = 196 regions
  {
    const float* s = si + b * R_DIM;
    const float x0 = (tid < R_DIM) ? s[tid] : -3.4e38f;
    const float mx = blk_max(x0, red, tid);
    const float e0 = (tid < R_DIM) ? __expf(x0 - mx) : 0.0f;
    const float sm = blk_sum(e0, red, tid);
    if (tid < R_DIM) p[tid] = e0 / sm;
    __syncthreads();
    for (int d = tid; d < D_DIM; d += 256) {
      const float* base = img + (size_t)b * R_DIM * D_DIM + d;
      float acc = 0.0f;
      for (int r = 0; r < R_DIM; ++r) acc += p[r] * base[(size_t)r * D_DIM];
      v1[b * D_DIM + d] = acc;
    }
  }
  __syncthreads();

  // ---- textual attention over S = 512 tokens
  {
    const float* s = kt + b * S_DIM;
    const float x0 = s[tid];
    const float x1 = s[tid + 256];
    const float mx = blk_max(fmaxf(x0, x1), red, tid);
    const float e0 = __expf(x0 - mx);
    const float e1 = __expf(x1 - mx);
    const float sm = blk_sum(e0 + e1, red, tid);
    p[tid] = e0 / sm;
    p[tid + 256] = e1 / sm;
    __syncthreads();
    for (int d = tid; d < D_DIM; d += 256) {
      const float* base = text + (size_t)b * S_DIM * D_DIM + d;
      float acc = 0.0f;
      for (int j = 0; j < S_DIM; ++j) acc += p[j] * base[(size_t)j * D_DIM];
      v2[b * D_DIM + d] = acc;
    }
  }
}

// ---------------------------------------------------------------------------
// Kernel 4: broadcast pooled vectors to full outputs (float4 stores).
// out[0 .. B*S*D)       = att_text[b,i,d] = v2[b,d]
// out[B*S*D .. 2*B*S*D) = att_img [b,s,d] = v1[b,d]
// ---------------------------------------------------------------------------
__global__ __launch_bounds__(256)
void k_broadcast(const float* __restrict__ v1, const float* __restrict__ v2,
                 float* __restrict__ out) {
  const size_t total = (size_t)B_DIM * S_DIM * (D_DIM / 4);
  size_t idx = (size_t)blockIdx.x * 256 + threadIdx.x;   // float4 units
  if (idx >= total) return;
  const int dq = (int)(idx % (D_DIM / 4));
  const int b  = (int)(idx / ((size_t)S_DIM * (D_DIM / 4)));
  const float4* v1q = reinterpret_cast<const float4*>(v1);
  const float4* v2q = reinterpret_cast<const float4*>(v2);
  float4* o = reinterpret_cast<float4*>(out);
  o[idx]         = v2q[b * (D_DIM / 4) + dq];
  o[total + idx] = v1q[b * (D_DIM / 4) + dq];
}

// ---------------------------------------------------------------------------
extern "C" void kernel_launch(void* const* d_in, const int* in_sizes, int n_in,
                              void* d_out, int out_size, void* d_ws, size_t ws_size,
                              hipStream_t stream) {
  (void)in_sizes; (void)n_in; (void)out_size; (void)ws_size;

  const float* text = (const float*)d_in[0];   // (B,S,D)
  const float* img  = (const float*)d_in[1];   // (B,R,D)
  const float* Wi1  = (const float*)d_in[4];   // (D,D)
  const float* wa1  = (const float*)d_in[5];   // (2D)
  const float* Wt2  = (const float*)d_in[7];   // (D,D)
  const float* wa2  = (const float*)d_in[10];  // (2D)
  float* out = (float*)d_out;

  // workspace layout (256B-aligned offsets)
  char* ws = (char*)d_ws;
  __bf16* WT_i1 = (__bf16*)(ws);                         // 1,179,648 B
  __bf16* WT_t2 = (__bf16*)(ws + 1179648);               // 1,179,648 B
  float*  si    = (float*)(ws + 2359296);                // B*R f32
  float*  kt    = (float*)(ws + 2384384);                // B*S f32
  float*  v1    = (float*)(ws + 2449920);                // B*D f32
  float*  v2    = (float*)(ws + 2548224);                // B*D f32

  const int wblocks = (D_DIM * D_DIM + 255) / 256;       // 2304
  k_weightT<<<wblocks, 256, 0, stream>>>(Wi1, WT_i1);
  k_weightT<<<wblocks, 256, 0, stream>>>(Wt2, WT_t2);

  // si = tanh(img @ Wi1) . wa1[D:2D]   (M = B*R = 6272 -> 392 blocks)
  k_gemm_tanh_dot<<<(B_DIM * R_DIM) / 16, 256, GEMM_LDS_BYTES, stream>>>(
      img, WT_i1, wa1 + D_DIM, si, B_DIM * R_DIM);
  // kt = tanh(text @ Wt2) . wa2[D:2D]  (M = B*S = 16384 -> 1024 blocks)
  k_gemm_tanh_dot<<<(B_DIM * S_DIM) / 16, 256, GEMM_LDS_BYTES, stream>>>(
      text, WT_t2, wa2 + D_DIM, kt, B_DIM * S_DIM);

  k_attend_pool<<<B_DIM, 256, 0, stream>>>(si, kt, img, text, v1, v2);

  const size_t totq = (size_t)B_DIM * S_DIM * (D_DIM / 4);
  k_broadcast<<<(unsigned)((totq + 255) / 256), 256, 0, stream>>>(v1, v2, out);
}
